// MFB_reverse_52286931862171
// MI455X (gfx1250) — matrix operand order repaired and checked by reading, not compiled, not run on hardware
//
#include <hip/hip_runtime.h>

// ---------------------------------------------------------------------------
// MFB bilinear attention pipeline for MI455X (gfx1250, wave32, WMMA).
// Shapes: B=32, T1=T2=512, D1=D2=O=256, KF=8, G=64.
// Dominant cost: F[b,i,j] = sum_k (Uk[k,i,:]. R[b,:,j]) * (Vk[k,i,:]. Q[b,:,j])
// (~69 GFLOP) -> fused per-tile dual-WMMA kernel, f16 inputs / f32 accum.
// ---------------------------------------------------------------------------

typedef __attribute__((ext_vector_type(16))) _Float16 v16h;
typedef __attribute__((ext_vector_type(8)))  _Float16 v8h;
typedef __attribute__((ext_vector_type(8)))  float    v8f;

#define NEGC 1e9f

// --- WMMA fragment loaders (per CDNA5 ISA VGPR layouts, wave32) -------------
// A (16x32 f16): lane L (L<16): row=L, K 0..7 & 16..23; L>=16: row=L-16, K 8..15 & 24..31
__device__ __forceinline__ v16h frag_a(const _Float16* __restrict__ lds, int row0, int ld) {
  const int lane = threadIdx.x & 31;
  const _Float16* p = lds + (row0 + (lane & 15)) * ld + ((lane >> 4) << 3);
  v8h lo = *(const v8h*)p;
  v8h hi = *(const v8h*)(p + 16);
  v16h a;
#pragma unroll
  for (int i = 0; i < 8; ++i) { a[i] = lo[i]; a[i + 8] = hi[i]; }
  return a;
}
// B (32x16 f16): lane L: col=L&15, K (L>>4)*8 .. +7 & 16+(L>>4)*8 .. +7. Tile staged transposed
// in LDS as BT[n][k] (k contiguous) so this is 2 contiguous 16B ds loads.
__device__ __forceinline__ v16h frag_b(const _Float16* __restrict__ lds, int col0, int ld) {
  const int lane = threadIdx.x & 31;
  const _Float16* p = lds + (col0 + (lane & 15)) * ld + ((lane >> 4) << 3);
  v8h lo = *(const v8h*)p;
  v8h hi = *(const v8h*)(p + 16);
  v16h b;
#pragma unroll
  for (int i = 0; i < 8; ++i) { b[i] = lo[i]; b[i + 8] = hi[i]; }
  return b;
}
__device__ __forceinline__ v8f wmma(v16h a, v16h b, v8f c) {
  return __builtin_amdgcn_wmma_f32_16x16x32_f16(false, a, false, b, (short)0, c, false, false);
}
// Convert 16 contiguous f32 -> 16 contiguous f16 in LDS (2x ds_store_b128).
__device__ __forceinline__ void cvt16_store(const float* __restrict__ src,
                                            _Float16* __restrict__ dst) {
  v8h a, b;
#pragma unroll
  for (int j = 0; j < 8; ++j) { a[j] = (_Float16)src[j]; b[j] = (_Float16)src[j + 8]; }
  *(v8h*)dst = a;
  *(v8h*)(dst + 8) = b;
}
// Copy 16 contiguous f16 global -> LDS (b128 load + ds_store_b128).
__device__ __forceinline__ void cp16h(const _Float16* __restrict__ src,
                                      _Float16* __restrict__ dst) {
  v8h a = *(const v8h*)src;
  v8h b = *(const v8h*)(src + 8);
  *(v8h*)dst = a;
  *(v8h*)(dst + 8) = b;
}

// --- K1: R_l / Q_l projections: C(256x512) = W(256x256) @ X(256x512), b-wise.
// in1 = x0.reshape(B,D,T) is a flat reinterpret -> X row-major (d,t), t contig.
// Epilogue: relu(C - NEG*pad + bias). Tile 128x128, 8 waves of 32x64.
__global__ __launch_bounds__(256) void k_proj(
    const float* __restrict__ x0, const float* __restrict__ x1,
    const float* __restrict__ W_R, const float* __restrict__ W_Q,
    const float* __restrict__ br, const float* __restrict__ bq,
    const unsigned char* __restrict__ q_mask, const unsigned char* __restrict__ v_mask,
    float* __restrict__ Rl, float* __restrict__ Ql)
{
  const int which = blockIdx.z & 1;
  const int b = blockIdx.z >> 1;
  const int n0 = blockIdx.x * 128;   // t
  const int m0 = blockIdx.y * 128;   // o
  const float* W = which ? W_Q : W_R;
  const float* X = (which ? x1 : x0) + (size_t)b * 256 * 512;
  const float* bias = which ? bq : br;
  const unsigned char* msk = (which ? v_mask : q_mask) + (size_t)b * 512;
  float* out = (which ? Ql : Rl) + (size_t)b * 256 * 512;

  __shared__ __align__(16) _Float16 As[128 * 32];
  __shared__ __align__(16) _Float16 Bs[128 * 32];   // transposed [n][k]

  const int tid = threadIdx.x, lane = tid & 31, wid = tid >> 5;
  const int wm = wid & 3, wn = wid >> 2;

  v8f zero = {};
  v8f acc[2][4];
#pragma unroll
  for (int i = 0; i < 2; ++i)
#pragma unroll
    for (int c = 0; c < 4; ++c) acc[i][c] = zero;

  for (int k0 = 0; k0 < 256; k0 += 32) {
    { int row = tid >> 1, ko = (tid & 1) * 16;
      cvt16_store(W + (size_t)(m0 + row) * 256 + k0 + ko, As + row * 32 + ko); }
    { int kk = tid >> 3, nb = (tid & 7) * 16;
      const float* src = X + (size_t)(k0 + kk) * 512 + n0 + nb;
#pragma unroll
      for (int j = 0; j < 16; ++j) Bs[(nb + j) * 32 + kk] = (_Float16)src[j]; }
    __syncthreads();
    v16h a0 = frag_a(As, wm * 32, 32);
    v16h a1 = frag_a(As, wm * 32 + 16, 32);
#pragma unroll
    for (int c = 0; c < 4; ++c) {
      v16h bm = frag_b(Bs, wn * 64 + c * 16, 32);
      acc[0][c] = wmma(a0, bm, acc[0][c]);
      acc[1][c] = wmma(a1, bm, acc[1][c]);
    }
    __syncthreads();
  }
  const int hi = lane >> 4, ln = lane & 15;
#pragma unroll
  for (int i = 0; i < 2; ++i)
#pragma unroll
    for (int c = 0; c < 4; ++c) {
      int tcol = n0 + wn * 64 + c * 16 + ln;
      float mterm = msk[tcol] ? 0.f : -NEGC;
#pragma unroll
      for (int r = 0; r < 8; ++r) {
        int o = m0 + wm * 32 + i * 16 + hi * 8 + r;
        float v = acc[i][c][r] + mterm + bias[o];
        out[(size_t)o * 512 + tcol] = v > 0.f ? v : 0.f;
      }
    }
}

// --- K2: pack Uk/Vk to f16, layout [k][i][o] (o contiguous = GEMM K-dim).
// Uk[k,i,o] = U0[((i*256+o)>>9), ((i*256+o)&511), k]  (flat-reshape gather)
__global__ __launch_bounds__(256) void k_pack_uv(
    const float* __restrict__ U0, const float* __restrict__ V0,
    _Float16* __restrict__ Up, _Float16* __restrict__ Vp)
{
  const int idx = blockIdx.x * 256 + threadIdx.x;    // < 8*512*256
  const float* src = blockIdx.y ? V0 : U0;
  _Float16* dst = blockIdx.y ? Vp : Up;
  const int o = idx & 255, i = (idx >> 8) & 511, k = idx >> 17;
  const int e = i * 256 + o, q = e >> 9, r = e & 511;
  dst[idx] = (_Float16)src[q * 4096 + r * 8 + k];
}

// --- K3: fused bilinear F (dominant).  Tile 64x64, 4 waves of 16x64.
// For each k: P = Uk[i,:]@R[:,j], S = Vk[i,:]@Q[:,j] (K=256), F += P.*S.
__global__ __launch_bounds__(128) void k_bilinear_f(
    const _Float16* __restrict__ Up, const _Float16* __restrict__ Vp,
    const float* __restrict__ Rl, const float* __restrict__ Ql,
    float* __restrict__ F)
{
  const int j0 = blockIdx.x * 64;
  const int i0 = blockIdx.y * 64;
  const int b = blockIdx.z;
  const float* R = Rl + (size_t)b * 256 * 512;
  const float* Q = Ql + (size_t)b * 256 * 512;
  float* Fb = F + (size_t)b * 512 * 512;

  __shared__ __align__(16) _Float16 Au[64 * 32], Av[64 * 32];
  __shared__ __align__(16) _Float16 Brt[64 * 32], Bqt[64 * 32];

  const int tid = threadIdx.x, lane = tid & 31, w = tid >> 5;
  v8f zero = {};
  v8f Fa[4];
#pragma unroll
  for (int c = 0; c < 4; ++c) Fa[c] = zero;

  for (int k = 0; k < 8; ++k) {
    const _Float16* Uk = Up + (size_t)k * 131072;
    const _Float16* Vk = Vp + (size_t)k * 131072;
    v8f P[4], S[4];
#pragma unroll
    for (int c = 0; c < 4; ++c) { P[c] = zero; S[c] = zero; }
    for (int o0 = 0; o0 < 256; o0 += 32) {
      { int row = tid >> 1, ko = (tid & 1) * 16;
        const size_t goff = (size_t)(i0 + row) * 256 + o0 + ko;
        cp16h(Uk + goff, Au + row * 32 + ko);
        cp16h(Vk + goff, Av + row * 32 + ko); }
      { int kk = tid >> 2, nb = (tid & 3) * 16;
        const float* sr = R + (size_t)(o0 + kk) * 512 + j0 + nb;
        const float* sq = Q + (size_t)(o0 + kk) * 512 + j0 + nb;
#pragma unroll
        for (int j = 0; j < 16; ++j) {
          Brt[(nb + j) * 32 + kk] = (_Float16)sr[j];
          Bqt[(nb + j) * 32 + kk] = (_Float16)sq[j];
        } }
      __syncthreads();
      v16h au = frag_a(Au, w * 16, 32);
      v16h av = frag_a(Av, w * 16, 32);
#pragma unroll
      for (int c = 0; c < 4; ++c) {
        v16h bmr = frag_b(Brt, c * 16, 32);
        P[c] = wmma(au, bmr, P[c]);
        v16h bmq = frag_b(Bqt, c * 16, 32);
        S[c] = wmma(av, bmq, S[c]);
      }
      __syncthreads();
    }
#pragma unroll
    for (int c = 0; c < 4; ++c)
#pragma unroll
      for (int r = 0; r < 8; ++r) Fa[c][r] += P[c][r] * S[c][r];
  }
  const int hi = lane >> 4, ln = lane & 15;
#pragma unroll
  for (int c = 0; c < 4; ++c)
#pragma unroll
    for (int r = 0; r < 8; ++r) {
      int ii = i0 + w * 16 + hi * 8 + r;
      int jj = j0 + c * 16 + ln;
      Fb[(size_t)ii * 512 + jj] = Fa[c][r];
    }
}

// --- K4: WR = W2_R(64x256) @ R_l[b], WQ = W2_Q @ Q_l[b]. Tile 64x128.
__global__ __launch_bounds__(128) void k_w2(
    const float* __restrict__ W2R, const float* __restrict__ W2Q,
    const float* __restrict__ Rl, const float* __restrict__ Ql,
    float* __restrict__ WR, float* __restrict__ WQ)
{
  const int which = blockIdx.z & 1;
  const int b = blockIdx.z >> 1;
  const int n0 = blockIdx.x * 128;
  const float* W2 = which ? W2Q : W2R;
  const float* X = (which ? Ql : Rl) + (size_t)b * 256 * 512;
  float* out = (which ? WQ : WR) + (size_t)b * 64 * 512;

  __shared__ __align__(16) _Float16 As[64 * 32];
  __shared__ __align__(16) _Float16 Bt[128 * 32];

  const int tid = threadIdx.x, lane = tid & 31, w = tid >> 5;
  v8f zero = {};
  v8f acc[8];
#pragma unroll
  for (int c = 0; c < 8; ++c) acc[c] = zero;

  for (int k0 = 0; k0 < 256; k0 += 32) {
    { int row = tid >> 1, ko = (tid & 1) * 16;
      cvt16_store(W2 + (size_t)row * 256 + k0 + ko, As + row * 32 + ko); }
    { int kk = tid >> 2, nb = (tid & 3) * 32;
      const float* src = X + (size_t)(k0 + kk) * 512 + n0 + nb;
#pragma unroll
      for (int j = 0; j < 32; ++j) Bt[(nb + j) * 32 + kk] = (_Float16)src[j]; }
    __syncthreads();
    v16h a = frag_a(As, w * 16, 32);
#pragma unroll
    for (int c = 0; c < 8; ++c) {
      v16h bm = frag_b(Bt, c * 16, 32);
      acc[c] = wmma(a, bm, acc[c]);
    }
    __syncthreads();
  }
  const int hi = lane >> 4, ln = lane & 15;
#pragma unroll
  for (int c = 0; c < 8; ++c)
#pragma unroll
    for (int r = 0; r < 8; ++r) {
      int g = w * 16 + hi * 8 + r;
      int t = n0 + c * 16 + ln;
      out[(size_t)g * 512 + t] = acc[c][r];
    }
}

// --- K5: M_v/M_q GEMM fused with relu and weighted-g reduction -> d1/d2.
// M = relu(base + Wx @ F[b]); d[t] = sum_g w[g]*M[g,t] - NEG*pad[t].
// Tile 64(all G) x 128; M never stored. Deterministic shuffle+LDS reduction.
__global__ __launch_bounds__(128) void k_md(
    const float* __restrict__ WR, const float* __restrict__ WQ,
    const float* __restrict__ F,
    const float* __restrict__ w_mv, const float* __restrict__ w_mq,
    const unsigned char* __restrict__ q_mask, const unsigned char* __restrict__ v_mask,
    float* __restrict__ d1, float* __restrict__ d2)
{
  const int which = blockIdx.z & 1;
  const int b = blockIdx.z >> 1;
  const int n0 = blockIdx.x * 128;
  const float* base = (which ? WQ : WR) + (size_t)b * 64 * 512;
  const float* A = (which ? WR : WQ) + (size_t)b * 64 * 512;
  const float* Fb = F + (size_t)b * 512 * 512;
  const float* wv = which ? w_mq : w_mv;
  const unsigned char* msk = (which ? v_mask : q_mask) + (size_t)b * 512;
  float* dd = (which ? d2 : d1) + (size_t)b * 512;

  __shared__ __align__(16) _Float16 As[64 * 32];
  __shared__ __align__(16) _Float16 Bt[128 * 32];
  __shared__ float wcol[4][128];

  const int tid = threadIdx.x, lane = tid & 31, w = tid >> 5;
  v8f zero = {};
  v8f acc[8];
#pragma unroll
  for (int c = 0; c < 8; ++c) acc[c] = zero;

  for (int k0 = 0; k0 < 512; k0 += 32) {
    { int row = tid >> 1, ko = (tid & 1) * 16;
      cvt16_store(A + (size_t)row * 512 + k0 + ko, As + row * 32 + ko); }
    { int kk = tid >> 2, nb = (tid & 3) * 32;
      const float* src = Fb + (size_t)(k0 + kk) * 512 + n0 + nb;
#pragma unroll
      for (int j = 0; j < 32; ++j) Bt[(nb + j) * 32 + kk] = (_Float16)src[j]; }
    __syncthreads();
    v16h a = frag_a(As, w * 16, 32);
#pragma unroll
    for (int c = 0; c < 8; ++c) {
      v16h bm = frag_b(Bt, c * 16, 32);
      acc[c] = wmma(a, bm, acc[c]);
    }
    __syncthreads();
  }
  const int hi = lane >> 4, ln = lane & 15;
#pragma unroll
  for (int c = 0; c < 8; ++c) {
    float part = 0.f;
#pragma unroll
    for (int r = 0; r < 8; ++r) {
      int g = w * 16 + hi * 8 + r;
      int t = n0 + c * 16 + ln;
      float v = base[(size_t)g * 512 + t] + acc[c][r];
      v = v > 0.f ? v : 0.f;                 // relu -> M element
      part += wv[g] * v;
    }
    part += __shfl_down(part, 16, 32);       // combine row-halves per column
    if (lane < 16) wcol[w][c * 16 + ln] = part;
  }
  __syncthreads();
  if (tid < 128) {
    float s = wcol[0][tid] + wcol[1][tid] + wcol[2][tid] + wcol[3][tid];
    int t = n0 + tid;
    dd[t] = s + (msk[t] ? 0.f : -NEGC);
  }
}

// --- K6: row softmax over T=512 -> gamma.
__global__ __launch_bounds__(256) void k_softmax(
    const float* __restrict__ d1, const float* __restrict__ d2,
    float* __restrict__ g1, float* __restrict__ g2)
{
  const int b = blockIdx.x;
  const float* d = (blockIdx.y ? d2 : d1) + (size_t)b * 512;
  float* g = (blockIdx.y ? g2 : g1) + (size_t)b * 512;
  __shared__ float red[256];
  const int tid = threadIdx.x;
  float a = d[tid], c = d[tid + 256];
  red[tid] = fmaxf(a, c);
  __syncthreads();
  for (int s = 128; s > 0; s >>= 1) { if (tid < s) red[tid] = fmaxf(red[tid], red[tid + s]); __syncthreads(); }
  float mx = red[0];
  __syncthreads();
  float e0 = expf(a - mx), e1 = expf(c - mx);
  red[tid] = e0 + e1;
  __syncthreads();
  for (int s = 128; s > 0; s >>= 1) { if (tid < s) red[tid] += red[tid + s]; __syncthreads(); }
  float inv = 1.f / red[0];
  g[tid] = e0 * inv;
  g[tid + 256] = e1 * inv;
}

// --- K7: trace/log embeddings and output. R_l2 reshape = same flat buffer:
// R_l2[b,t,o] = Rl_flat[b*131072 + t*256 + o]  (coalesced over o).
__global__ __launch_bounds__(256) void k_out(
    const float* __restrict__ Rl, const float* __restrict__ Ql,
    const float* __restrict__ g1, const float* __restrict__ g2,
    float* __restrict__ out)
{
  const int b = blockIdx.x;
  const int o = threadIdx.x;  // 0..255
  __shared__ float gv[512], gq[512];
  for (int t = threadIdx.x; t < 512; t += 256) { gv[t] = g1[b * 512 + t]; gq[t] = g2[b * 512 + t]; }
  __syncthreads();
  const float* R = Rl + (size_t)b * 131072;
  const float* Q = Ql + (size_t)b * 131072;
  float tr = 0.f, lg = 0.f;
  for (int t = 0; t < 512; ++t) {
    tr += gv[t] * R[t * 256 + o];
    lg += gq[t] * Q[t * 256 + o];
  }
  out[b * 256 + o] = tr * lg;
}

extern "C" void kernel_launch(void* const* d_in, const int* in_sizes, int n_in,
                              void* d_out, int out_size, void* d_ws, size_t ws_size,
                              hipStream_t stream) {
  const float* x0 = (const float*)d_in[0];
  const float* x1 = (const float*)d_in[1];
  const unsigned char* q_mask = (const unsigned char*)d_in[2];  // jnp bool = 1 byte
  const unsigned char* v_mask = (const unsigned char*)d_in[3];
  const float* W_R  = (const float*)d_in[4];
  const float* W_Q  = (const float*)d_in[5];
  const float* br   = (const float*)d_in[6];
  const float* bq   = (const float*)d_in[7];
  const float* U0   = (const float*)d_in[8];
  const float* V0   = (const float*)d_in[9];
  const float* W2R  = (const float*)d_in[10];
  const float* W2Q  = (const float*)d_in[11];
  const float* w_mv = (const float*)d_in[12];
  const float* w_mq = (const float*)d_in[13];

  char* ws = (char*)d_ws;
  float* Rl = (float*)ws;       ws += (size_t)32 * 256 * 512 * 4;   // 16 MB
  float* Ql = (float*)ws;       ws += (size_t)32 * 256 * 512 * 4;   // 16 MB
  _Float16* Up = (_Float16*)ws; ws += (size_t)8 * 512 * 256 * 2;    // 2 MB
  _Float16* Vp = (_Float16*)ws; ws += (size_t)8 * 512 * 256 * 2;    // 2 MB
  float* F  = (float*)ws;       ws += (size_t)32 * 512 * 512 * 4;   // 32 MB
  float* WR = (float*)ws;       ws += (size_t)32 * 64 * 512 * 4;    // 4 MB
  float* WQ = (float*)ws;       ws += (size_t)32 * 64 * 512 * 4;    // 4 MB
  float* d1 = (float*)ws;       ws += (size_t)32 * 512 * 4;
  float* d2 = (float*)ws;       ws += (size_t)32 * 512 * 4;
  float* g1 = (float*)ws;       ws += (size_t)32 * 512 * 4;
  float* g2 = (float*)ws;       ws += (size_t)32 * 512 * 4;

  k_proj<<<dim3(4, 2, 64), 256, 0, stream>>>(x0, x1, W_R, W_Q, br, bq, q_mask, v_mask, Rl, Ql);
  k_pack_uv<<<dim3(4096, 2), 256, 0, stream>>>(U0, V0, Up, Vp);
  k_bilinear_f<<<dim3(8, 8, 32), 128, 0, stream>>>(Up, Vp, Rl, Ql, F);
  k_w2<<<dim3(4, 1, 64), 128, 0, stream>>>(W2R, W2Q, Rl, Ql, WR, WQ);
  k_md<<<dim3(4, 1, 64), 128, 0, stream>>>(WR, WQ, F, w_mv, w_mq, q_mask, v_mask, d1, d2);
  k_softmax<<<dim3(32, 2), 256, 0, stream>>>(d1, d2, g1, g2);
  k_out<<<32, 256, 0, stream>>>(Rl, Ql, g1, g2, (float*)d_out);
}
